// HyperNetwork_GINO_21208548507863
// MI455X (gfx1250) — matrix operand order repaired and checked by reading, compiled, not run on hardware
//
#include <hip/hip_runtime.h>
#include <math.h>

// ---------------------------------------------------------------------------
// CDNA5 (gfx1250) wave32 WMMA types / helpers
// ---------------------------------------------------------------------------
typedef __attribute__((ext_vector_type(2)))  _Float16 h2;
typedef __attribute__((ext_vector_type(16))) _Float16 v16h;
typedef __attribute__((ext_vector_type(8)))  float    v8f;

#define WMMA_F16(a, b, c) \
  __builtin_amdgcn_wmma_f32_16x16x32_f16(false, (a), false, (b), (short)0, (c), false, false)

// A fragment (16x32 f16, MxK): lane<16 row=lane holds K {0..7,16..23};
// lane>=16 row=lane-16 holds K {8..15,24..31}.  `row` = LDS row ptr (K-major).
__device__ __forceinline__ v16h frag_a(const _Float16* row, int half) {
  v16h a;
#pragma unroll
  for (int j = 0; j < 4; ++j) {
    h2 p = *(const h2*)(row + 8 * half + 2 * j);
    a[2 * j] = p[0]; a[2 * j + 1] = p[1];
  }
#pragma unroll
  for (int j = 0; j < 4; ++j) {
    h2 p = *(const h2*)(row + 16 + 8 * half + 2 * j);
    a[8 + 2 * j] = p[0]; a[9 + 2 * j] = p[1];
  }
  return a;
}

// B fragment (32x16 f16, KxN): lane<16 col=lane holds K {0..7,16..23}; lane>=16 K {8..15,24..31}.
__device__ __forceinline__ v16h frag_b(const _Float16* col, int half) {
  v16h b;
#pragma unroll
  for (int j = 0; j < 8; ++j) {
    h2 p = *(const h2*)(col + ((j & 4) << 2) + 8 * half + 2 * (j & 3));
    b[2 * j] = p[0]; b[2 * j + 1] = p[1];
  }
  return b;
}

// ---------------------------------------------------------------------------
// Generic WMMA GEMM: C[M,N] = A[M,K] @ W[K,N] + bias, with epilogues.
// 128 threads = 4 waves, block tile 64x64, each wave 32x32 (2x2 WMMA tiles).
// Double-buffered LDS: global loads for step s+1 overlap WMMAs of step s,
// global_prefetch issued for step s+2.
// ---------------------------------------------------------------------------
constexpr int EPI_NONE = 0;
constexpr int EPI_GELU = 1;        // tanh-approx GELU
constexpr int EPI_RESID_GATE = 2;  // C = resid + gate[b]*(acc+bias)
constexpr int EPI_POSEMB = 3;      // C = acc+bias + DiT 2D sin-cos pos-embed

#define TM 64
#define TN 64
#define TK 32

template <int EPI>
__launch_bounds__(128)
__global__ void gemm_kernel(const float* __restrict__ A, const float* __restrict__ W,
                            const float* __restrict__ bias, float* __restrict__ C,
                            int M, int N, int K,
                            const float* __restrict__ resid,
                            const float* __restrict__ gate, int gate_stride, int rows_per_b) {
  __shared__ _Float16 As[2][TM][TK + 2];  // K-major per row
  __shared__ _Float16 Bs[2][TN][TK + 2];  // Bs[.][n][k] = W[k][n]
  const int tid = threadIdx.x;
  const int lane = tid & 31, wave = tid >> 5;
  const int half = lane >> 4, l16 = lane & 15;
  const int bm = blockIdx.y * TM, bn = blockIdx.x * TN;
  const int wm = (wave >> 1) * 32, wn = (wave & 1) * 32;
  const int nk = (K + TK - 1) / TK;
  const int rb_l2 = 31 - __clz(rows_per_b);  // rows_per_b is a power of two

  float pa[16], pb[16];
  auto load_tile = [&](int k0) {
#pragma unroll
    for (int u = 0; u < 16; ++u) {
      int i = tid + u * 128;
      int r = i >> 5, c = i & 31;
      int gm = bm + r, gk = k0 + c;
      pa[u] = (gm < M && gk < K) ? A[(size_t)gm * K + gk] : 0.0f;
      int gn = bn + r;
      pb[u] = (gk < K && gn < N) ? W[(size_t)gk * N + gn] : 0.0f;
    }
  };
  auto store_tile = [&](int buf) {
#pragma unroll
    for (int u = 0; u < 16; ++u) {
      int i = tid + u * 128;
      int r = i >> 5, c = i & 31;
      As[buf][r][c] = (_Float16)pa[u];
      Bs[buf][r][c] = (_Float16)pb[u];
    }
  };

  v8f acc[2][2] = {};
  load_tile(0);
  store_tile(0);
  __syncthreads();
  int cur = 0;
  for (int s = 0; s < nk; ++s) {
    if (s + 1 < nk) load_tile((s + 1) * TK);       // overlap with WMMAs below
    if (s + 2 < nk) {                              // L2 prefetch two steps ahead
      int i = tid, r = i >> 5, c = i & 31;
      int gm = bm + r, gk = (s + 2) * TK + c;
      if (gm < M && gk < K) __builtin_prefetch(&A[(size_t)gm * K + gk], 0, 3);
      int gn = bn + r;
      if (gk < K && gn < N) __builtin_prefetch(&W[(size_t)gk * N + gn], 0, 3);
    }
    v16h af[2], bf[2];
#pragma unroll
    for (int mi = 0; mi < 2; ++mi) af[mi] = frag_a(&As[cur][wm + 16 * mi + l16][0], half);
#pragma unroll
    for (int ni = 0; ni < 2; ++ni) bf[ni] = frag_b(&Bs[cur][wn + 16 * ni + l16][0], half);
#pragma unroll
    for (int mi = 0; mi < 2; ++mi)
#pragma unroll
      for (int ni = 0; ni < 2; ++ni) acc[mi][ni] = WMMA_F16(af[mi], bf[ni], acc[mi][ni]);
    if (s + 1 < nk) {
      store_tile(cur ^ 1);
      __syncthreads();
      cur ^= 1;
    }
  }

#pragma unroll
  for (int mi = 0; mi < 2; ++mi)
#pragma unroll
    for (int ni = 0; ni < 2; ++ni) {
      int col = bn + wn + 16 * ni + l16;
      if (col >= N) continue;
      float bv = bias ? bias[col] : 0.0f;
#pragma unroll
      for (int r = 0; r < 8; ++r) {
        int row = bm + wm + 16 * mi + r + 8 * half;
        if (row >= M) continue;
        float v = acc[mi][ni][r] + bv;
        if (EPI == EPI_GELU) {
          float u = 0.7978845608f * (v + 0.044715f * v * v * v);
          v = 0.5f * v * (1.0f + tanhf(u));
        } else if (EPI == EPI_RESID_GATE) {
          int b = row >> rb_l2;
          v = resid[(size_t)row * N + col] + gate[b * gate_stride + col] * v;
        } else if (EPI == EPI_POSEMB) {
          int token = row & (rows_per_b - 1);  // rows_per_b == 1024
          int rr = token >> 5, cc = token & 31;
          int d = col;
          float pe;
          if (d < 128) {
            int i = d & 63;
            float omega = __expf(-9.210340372f * (float)i * (1.0f / 64.0f));
            float ang = (float)cc * omega;
            pe = (d < 64) ? __sinf(ang) : __cosf(ang);
          } else {
            int i = (d - 128) & 63;
            float omega = __expf(-9.210340372f * (float)i * (1.0f / 64.0f));
            float ang = (float)rr * omega;
            pe = (d < 192) ? __sinf(ang) : __cosf(ang);
          }
          v += pe;
        }
        C[(size_t)row * N + col] = v;
      }
    }
}

// ---------------------------------------------------------------------------
// Implicit-GEMM 3x3 conv (pad 1), NCHW in, double-buffered like gemm_kernel.
// OH, OW are powers of two at all call sites -> shift/mask addressing.
// ---------------------------------------------------------------------------
__launch_bounds__(128)
__global__ void conv3x3_kernel(const float* __restrict__ in, const float* __restrict__ w,
                               const float* __restrict__ bias, float* __restrict__ out,
                               int Bn, int Cin, int H, int Wd, int Cout, int OH, int OW,
                               int stride, int out_nhwc) {
  __shared__ _Float16 As[2][TM][TK + 2];
  __shared__ _Float16 Bs[2][TN][TK + 2];
  const int tid = threadIdx.x;
  const int lane = tid & 31, wave = tid >> 5;
  const int half = lane >> 4, l16 = lane & 15;
  const int M = Bn * OH * OW;
  const int bm = blockIdx.y * TM, bn = blockIdx.x * TN;
  const int wm = (wave >> 1) * 32, wn = (wave & 1) * 32;
  const int ow_l2 = 31 - __clz(OW);
  const int ohow_l2 = 31 - __clz(OH * OW);
  const int icsteps = Cin >> 5;            // Cin / 32
  const int nk = 9 * icsteps;

  float pa[16], pb[16];
  auto load_tile = [&](int s) {
    int tap = s / icsteps;                 // uniform scalar division
    int ic0 = (s - tap * icsteps) << 5;
    int ky = tap / 3, kx = tap - ky * 3;   // uniform
#pragma unroll
    for (int u = 0; u < 16; ++u) {
      int i = tid + u * 128;
      int r = i >> 5, c = i & 31;
      int m = bm + r;
      float v = 0.0f;
      if (m < M) {
        int b = m >> ohow_l2;
        int rem = m & ((OH * OW) - 1);
        int oy = rem >> ow_l2, ox = rem & (OW - 1);
        int iy = oy * stride + ky - 1;
        int ix = ox * stride + kx - 1;
        if (iy >= 0 && iy < H && ix >= 0 && ix < Wd)
          v = in[(((size_t)b * Cin + ic0 + c) * H + iy) * Wd + ix];
      }
      pa[u] = v;
      int oc = bn + r;
      pb[u] = (oc < Cout) ? w[(((size_t)oc * Cin + ic0 + c) * 3 + ky) * 3 + kx] : 0.0f;
    }
  };
  auto store_tile = [&](int buf) {
#pragma unroll
    for (int u = 0; u < 16; ++u) {
      int i = tid + u * 128;
      int r = i >> 5, c = i & 31;
      As[buf][r][c] = (_Float16)pa[u];
      Bs[buf][r][c] = (_Float16)pb[u];
    }
  };

  v8f acc[2][2] = {};
  load_tile(0);
  store_tile(0);
  __syncthreads();
  int cur = 0;
  for (int s = 0; s < nk; ++s) {
    if (s + 1 < nk) load_tile(s + 1);
    v16h af[2], bf[2];
#pragma unroll
    for (int mi = 0; mi < 2; ++mi) af[mi] = frag_a(&As[cur][wm + 16 * mi + l16][0], half);
#pragma unroll
    for (int ni = 0; ni < 2; ++ni) bf[ni] = frag_b(&Bs[cur][wn + 16 * ni + l16][0], half);
#pragma unroll
    for (int mi = 0; mi < 2; ++mi)
#pragma unroll
      for (int ni = 0; ni < 2; ++ni) acc[mi][ni] = WMMA_F16(af[mi], bf[ni], acc[mi][ni]);
    if (s + 1 < nk) {
      store_tile(cur ^ 1);
      __syncthreads();
      cur ^= 1;
    }
  }

#pragma unroll
  for (int mi = 0; mi < 2; ++mi)
#pragma unroll
    for (int ni = 0; ni < 2; ++ni) {
      int oc = bn + wn + 16 * ni + l16;
      if (oc >= Cout) continue;
      float bv = bias[oc];
#pragma unroll
      for (int r = 0; r < 8; ++r) {
        int m = bm + wm + 16 * mi + r + 8 * half;
        if (m >= M) continue;
        int b = m >> ohow_l2;
        int rem = m & ((OH * OW) - 1);
        int oy = rem >> ow_l2, ox = rem & (OW - 1);
        float v = acc[mi][ni][r] + bv;
        if (out_nhwc)
          out[(((size_t)b * OH + oy) * OW + ox) * Cout + oc] = v;
        else
          out[(((size_t)b * Cout + oc) * OH + oy) * OW + ox] = v;
      }
    }
}

// ---------------------------------------------------------------------------
// Fused flash attention. qkv: [B*T, 768] (q|k|v, head cols h*32..h*32+31).
// Block = 8 waves (256 thr), each wave owns 16 query rows. grid (T/128, B*H).
// ---------------------------------------------------------------------------
__launch_bounds__(256)
__global__ void attn_kernel(const float* __restrict__ qkv, float* __restrict__ o,
                            int Bn, int T, int Hn, int dh) {
  __shared__ _Float16 Kld[32][34];      // [token][d]
  __shared__ _Float16 Vt[32][34];       // [d][token]
  __shared__ _Float16 Pscr[8][16][34];  // per-wave P re-layout scratch
  const int bh = blockIdx.y;
  const int b = bh / Hn, h = bh % Hn;
  const int tid = threadIdx.x, lane = tid & 31, wave = tid >> 5;
  const int half = lane >> 4, l16 = lane & 15;
  const int q0 = blockIdx.x * 128 + wave * 16;
  const int D3 = 3 * Hn * dh;
  const float qscale = 0.17677669529663689f;  // 1/sqrt(32)

  v16h aq;
  {
    const float* qp = qkv + (size_t)(b * T + q0 + l16) * D3 + h * dh;
#pragma unroll
    for (int j = 0; j < 4; ++j) {
      int k = 8 * half + 2 * j;
      aq[2 * j]     = (_Float16)(qp[k] * qscale);
      aq[2 * j + 1] = (_Float16)(qp[k + 1] * qscale);
    }
#pragma unroll
    for (int j = 0; j < 4; ++j) {
      int k = 16 + 8 * half + 2 * j;
      aq[8 + 2 * j] = (_Float16)(qp[k] * qscale);
      aq[9 + 2 * j] = (_Float16)(qp[k + 1] * qscale);
    }
  }

  v8f o0 = {}, o1 = {};
  float mrow[8], lrow[8];
#pragma unroll
  for (int r = 0; r < 8; ++r) { mrow[r] = -1e30f; lrow[r] = 0.0f; }

  for (int c0 = 0; c0 < T; c0 += 32) {
    __syncthreads();
    for (int i = tid; i < 32 * 32; i += 256) {
      int tt = i >> 5, d = i & 31;
      size_t rowb = (size_t)(b * T + c0 + tt) * D3 + h * dh;
      Kld[tt][d] = (_Float16)qkv[rowb + Hn * dh + d];
      Vt[d][tt]  = (_Float16)qkv[rowb + 2 * Hn * dh + d];
    }
    __syncthreads();

    v16h bk0 = frag_b(&Kld[l16][0], half);
    v16h bk1 = frag_b(&Kld[16 + l16][0], half);
    v8f z = {};
    v8f s0 = WMMA_F16(aq, bk0, z);
    v8f s1 = WMMA_F16(aq, bk1, z);

    float p0[8], p1[8];
#pragma unroll
    for (int r = 0; r < 8; ++r) {
      float mx = fmaxf(s0[r], s1[r]);
#pragma unroll
      for (int off = 8; off; off >>= 1) mx = fmaxf(mx, __shfl_xor(mx, off, 32));
      float mnew = fmaxf(mrow[r], mx);
      float alpha = __expf(mrow[r] - mnew);
      float e0 = __expf(s0[r] - mnew);
      float e1 = __expf(s1[r] - mnew);
      float sum = e0 + e1;
#pragma unroll
      for (int off = 8; off; off >>= 1) sum += __shfl_xor(sum, off, 32);
      lrow[r] = lrow[r] * alpha + sum;
      mrow[r] = mnew;
      o0[r] *= alpha; o1[r] *= alpha;
      p0[r] = e0; p1[r] = e1;
    }
#pragma unroll
    for (int r = 0; r < 8; ++r) {
      Pscr[wave][r + 8 * half][l16]      = (_Float16)p0[r];
      Pscr[wave][r + 8 * half][16 + l16] = (_Float16)p1[r];
    }
    // same-wave LDS ops are in order; compiler inserts the DScnt wait.
    v16h pa  = frag_a(&Pscr[wave][l16][0], half);
    v16h bv0 = frag_b(&Vt[l16][0], half);
    v16h bv1 = frag_b(&Vt[16 + l16][0], half);
    o0 = WMMA_F16(pa, bv0, o0);
    o1 = WMMA_F16(pa, bv1, o1);
  }

#pragma unroll
  for (int r = 0; r < 8; ++r) {
    float inv = 1.0f / lrow[r];
    size_t row = (size_t)(b * T + q0 + r + 8 * half) * (Hn * dh) + h * dh;
    o[row + l16]      = o0[r] * inv;
    o[row + 16 + l16] = o1[r] * inv;
  }
}

// ---------------------------------------------------------------------------
// Small kernels
// ---------------------------------------------------------------------------
// rfft(T=16, ortho) -> view_as_real features [B,N,36]
__global__ void fft_feats_kernel(const float* __restrict__ x, float* __restrict__ feats, int N) {
  int n = blockIdx.x * 256 + threadIdx.x;
  int b = blockIdx.y;
  if (n >= N) return;
  float xr[16][2];
#pragma unroll
  for (int t = 0; t < 16; ++t) {
    xr[t][0] = x[(((size_t)b * 16 + t) * N + n) * 2 + 0];
    xr[t][1] = x[(((size_t)b * 16 + t) * N + n) * 2 + 1];
  }
  float* out = feats + ((size_t)b * N + n) * 36;
  const float scale = 0.25f;  // 1/sqrt(16)
#pragma unroll
  for (int k = 0; k < 9; ++k) {
    float re0 = 0, im0 = 0, re1 = 0, im1 = 0;
#pragma unroll
    for (int t = 0; t < 16; ++t) {
      float ang = -(float)M_PI * (float)(k * t) * 0.125f;
      float cth = __cosf(ang), sth = __sinf(ang);
      re0 += xr[t][0] * cth; im0 += xr[t][0] * sth;
      re1 += xr[t][1] * cth; im1 += xr[t][1] * sth;
    }
    out[k * 4 + 0] = re0 * scale; out[k * 4 + 1] = im0 * scale;
    out[k * 4 + 2] = re1 * scale; out[k * 4 + 3] = im1 * scale;
  }
}

// top-3 nearest coords per latent-grid point, inverse-square weights
__launch_bounds__(256)
__global__ void knn_kernel(const float* __restrict__ coords, int* __restrict__ idx,
                           float* __restrict__ wts, int Mm, int Nn) {
  __shared__ float cs[256][2];
  int m = blockIdx.x * 256 + threadIdx.x;
  int b = blockIdx.y;
  int ix = m >> 6, iy = m & 63;
  float gx = -1.0f + 2.0f * (float)ix / 63.0f;
  float gy = -1.0f + 2.0f * (float)iy / 63.0f;
  float g2 = gx * gx + gy * gy;
  float bd0 = 1e30f, bd1 = 1e30f, bd2 = 1e30f;
  int bi0 = 0, bi1 = 0, bi2 = 0;
  for (int n0 = 0; n0 < Nn; n0 += 256) {
    __syncthreads();
    cs[threadIdx.x][0] = coords[((size_t)b * Nn + n0 + threadIdx.x) * 2 + 0];
    cs[threadIdx.x][1] = coords[((size_t)b * Nn + n0 + threadIdx.x) * 2 + 1];
    __syncthreads();
    for (int j = 0; j < 256; ++j) {
      float cx = cs[j][0], cy = cs[j][1];
      float d2 = g2 + cx * cx + cy * cy - 2.0f * (gx * cx + gy * cy);
      int n = n0 + j;
      if (d2 < bd0)      { bd2 = bd1; bi2 = bi1; bd1 = bd0; bi1 = bi0; bd0 = d2; bi0 = n; }
      else if (d2 < bd1) { bd2 = bd1; bi2 = bi1; bd1 = d2; bi1 = n; }
      else if (d2 < bd2) { bd2 = d2; bi2 = n; }
    }
  }
  float w0 = 1.0f / (fmaxf(bd0, 0.0f) + 1e-8f);
  float w1 = 1.0f / (fmaxf(bd1, 0.0f) + 1e-8f);
  float w2 = 1.0f / (fmaxf(bd2, 0.0f) + 1e-8f);
  float s = w0 + w1 + w2;
  size_t o = ((size_t)b * Mm + m) * 3;
  idx[o] = bi0; idx[o + 1] = bi1; idx[o + 2] = bi2;
  wts[o] = w0 / s; wts[o + 1] = w1 / s; wts[o + 2] = w2 / s;
}

__global__ void gather_kernel(const float* __restrict__ xh, const int* __restrict__ idx,
                              const float* __restrict__ wts, float* __restrict__ xg,
                              int Mm, int Nn, int D) {
  int bm = blockIdx.x;
  int b = bm / Mm;
  int c = threadIdx.x;
  size_t o = (size_t)bm * 3;
  float acc = 0.0f;
#pragma unroll
  for (int j = 0; j < 3; ++j)
    acc += wts[o + j] * xh[((size_t)b * Nn + idx[o + j]) * D + c];
  xg[(size_t)bm * D + c] = acc;
}

// timestep embedding + 2-layer MLP; stores silu(c)  [B,256]
__launch_bounds__(256)
__global__ void temb_kernel(const float* __restrict__ t, const float* __restrict__ w1,
                            const float* __restrict__ b1, const float* __restrict__ w2,
                            const float* __restrict__ b2, float* __restrict__ sc) {
  __shared__ float te[256], hh[256];
  int b = blockIdx.x, i = threadIdx.x;
  float tv = t[b];
  int ii = i & 127;
  float f = __expf(-9.210340372f * (float)ii * (1.0f / 128.0f));
  float ang = tv * f;
  te[i] = (i < 128) ? __cosf(ang) : __sinf(ang);
  __syncthreads();
  float acc = b1[i];
  for (int k = 0; k < 256; ++k) acc += te[k] * w1[k * 256 + i];
  hh[i] = acc / (1.0f + __expf(-acc));  // silu
  __syncthreads();
  float acc2 = b2[i];
  for (int k = 0; k < 256; ++k) acc2 += hh[k] * w2[k * 256 + i];
  sc[b * 256 + i] = acc2 / (1.0f + __expf(-acc2));  // silu(c)
}

// y[b][j] = x[b][:] @ W[:, j] + bias[j]   (tiny adaLN matvecs, Bn rows)
__global__ void matvec_kernel(const float* __restrict__ x, const float* __restrict__ W,
                              const float* __restrict__ bias, float* __restrict__ y,
                              int K, int N) {
  int j = blockIdx.x * 256 + threadIdx.x;
  int b = blockIdx.y;
  if (j >= N) return;
  float acc = bias[j];
  for (int k = 0; k < K; ++k) acc += x[b * K + k] * W[(size_t)k * N + j];
  y[(size_t)b * N + j] = acc;
}

// y = LN(x) * (1 + mod[b][off_scale+c]) + mod[b][off_shift+c]
__launch_bounds__(256)
__global__ void ln_mod_kernel(const float* __restrict__ x, float* __restrict__ y,
                              const float* __restrict__ mod, int mod_stride,
                              int off_scale, int off_shift, int rows_per_b_l2) {
  __shared__ float r1[256], r2[256];
  int row = blockIdx.x, c = threadIdx.x;
  float v = x[(size_t)row * 256 + c];
  r1[c] = v; r2[c] = v * v;
  __syncthreads();
  for (int o = 128; o; o >>= 1) {
    if (c < o) { r1[c] += r1[c + o]; r2[c] += r2[c + o]; }
    __syncthreads();
  }
  float mean = r1[0] * (1.0f / 256.0f);
  float var = r2[0] * (1.0f / 256.0f) - mean * mean;
  float inv = rsqrtf(var + 1e-6f);
  int b = row >> rows_per_b_l2;
  float scale = 1.0f + mod[(size_t)b * mod_stride + off_scale + c];
  float shift = mod[(size_t)b * mod_stride + off_shift + c];
  y[(size_t)row * 256 + c] = (v - mean) * inv * scale + shift;
}

// xp[b][t1*32+t2][c*4+p1*2+p2] = xg[b][(2t1+p1)*64 + (2t2+p2)][c]
__global__ void patchify_kernel(const float* __restrict__ xg, float* __restrict__ xp) {
  int bt = blockIdx.x;
  int b = bt >> 10, tokn = bt & 1023;
  int t1 = tokn >> 5, t2 = tokn & 31;
  int c = threadIdx.x;
#pragma unroll
  for (int p1 = 0; p1 < 2; ++p1)
#pragma unroll
    for (int p2 = 0; p2 < 2; ++p2) {
      int m = (2 * t1 + p1) * 64 + (2 * t2 + p2);
      xp[(size_t)bt * 1024 + c * 4 + p1 * 2 + p2] = xg[((size_t)b * 4096 + m) * 256 + c];
    }
}

// x2[b][c][2t1+p1][2t2+p2] = y[b][t1*32+t2][p1*512 + p2*256 + c]
__global__ void unpatchify_kernel(const float* __restrict__ y, float* __restrict__ x2) {
  int bt = blockIdx.x;
  int b = bt >> 10, tokn = bt & 1023;
  int t1 = tokn >> 5, t2 = tokn & 31;
  int c = threadIdx.x;
#pragma unroll
  for (int p1 = 0; p1 < 2; ++p1)
#pragma unroll
    for (int p2 = 0; p2 < 2; ++p2) {
      int ix = 2 * t1 + p1, iy = 2 * t2 + p2;
      x2[(((size_t)b * 256 + c) * 64 + ix) * 64 + iy] =
          y[(size_t)bt * 1024 + p1 * 512 + p2 * 256 + c];
    }
}

// GroupNorm(groups=8, eps 1e-5) + exact-erf GELU, NCHW. grid = B*groups.
__launch_bounds__(256)
__global__ void gn_gelu_kernel(const float* __restrict__ x, float* __restrict__ y,
                               const float* __restrict__ gw, const float* __restrict__ gb,
                               int C, int HW, int groups) {
  __shared__ float r1[256], r2[256];
  int bg = blockIdx.x;
  int g = bg % groups, b = bg / groups;
  int cpg = C / groups;
  size_t base = ((size_t)b * C + (size_t)g * cpg) * HW;
  int nelem = cpg * HW;
  float s = 0, s2 = 0;
  for (int i = threadIdx.x; i < nelem; i += 256) {
    float v = x[base + i];
    s += v; s2 += v * v;
  }
  r1[threadIdx.x] = s; r2[threadIdx.x] = s2;
  __syncthreads();
  for (int o = 128; o; o >>= 1) {
    if (threadIdx.x < o) { r1[threadIdx.x] += r1[threadIdx.x + o]; r2[threadIdx.x] += r2[threadIdx.x + o]; }
    __syncthreads();
  }
  float mean = r1[0] / (float)nelem;
  float var = r2[0] / (float)nelem - mean * mean;
  float inv = rsqrtf(var + 1e-5f);
  for (int i = threadIdx.x; i < nelem; i += 256) {
    int ch = g * cpg + i / HW;
    float v = (x[base + i] - mean) * inv * gw[ch] + gb[ch];
    y[base + i] = 0.5f * v * (1.0f + erff(v * 0.70710678f));
  }
}

// ---------------------------------------------------------------------------
// Launch
// ---------------------------------------------------------------------------
static inline dim3 gemm_grid(int M, int N) { return dim3((N + TN - 1) / TN, (M + TM - 1) / TM); }

extern "C" void kernel_launch(void* const* d_in, const int* in_sizes, int n_in,
                              void* d_out, int out_size, void* d_ws, size_t ws_size,
                              hipStream_t stream) {
  (void)in_sizes; (void)n_in; (void)out_size; (void)ws_size;
  // ---- unpack inputs (setup_inputs dict insertion order, recursive) ----
  int pi = 0;
  const float* x_noisy = (const float*)d_in[pi++];
  const float* coords  = (const float*)d_in[pi++];
  const float* t_in    = (const float*)d_in[pi++];
  const float* in_w    = (const float*)d_in[pi++];
  const float* in_b    = (const float*)d_in[pi++];
  const float* patch_w = (const float*)d_in[pi++];
  const float* patch_b = (const float*)d_in[pi++];
  const float* t_w1    = (const float*)d_in[pi++];
  const float* t_b1    = (const float*)d_in[pi++];
  const float* t_w2    = (const float*)d_in[pi++];
  const float* t_b2    = (const float*)d_in[pi++];
  struct Blk {
    const float *ada_w, *ada_b, *qkv_w, *qkv_b, *proj_w, *proj_b, *fc1_w, *fc1_b, *fc2_w, *fc2_b;
  } blk[4];
  for (int i = 0; i < 4; ++i) {
    blk[i].ada_w = (const float*)d_in[pi++]; blk[i].ada_b = (const float*)d_in[pi++];
    blk[i].qkv_w = (const float*)d_in[pi++]; blk[i].qkv_b = (const float*)d_in[pi++];
    blk[i].proj_w = (const float*)d_in[pi++]; blk[i].proj_b = (const float*)d_in[pi++];
    blk[i].fc1_w = (const float*)d_in[pi++]; blk[i].fc1_b = (const float*)d_in[pi++];
    blk[i].fc2_w = (const float*)d_in[pi++]; blk[i].fc2_b = (const float*)d_in[pi++];
  }
  const float* fl_ada_w = (const float*)d_in[pi++];
  const float* fl_ada_b = (const float*)d_in[pi++];
  const float* fl_w     = (const float*)d_in[pi++];
  const float* fl_b     = (const float*)d_in[pi++];
  const float* conv1_w  = (const float*)d_in[pi++];
  const float* conv1_b  = (const float*)d_in[pi++];
  const float* gn1_w    = (const float*)d_in[pi++];
  const float* gn1_b    = (const float*)d_in[pi++];
  const float* conv2_w  = (const float*)d_in[pi++];
  const float* conv2_b  = (const float*)d_in[pi++];
  const float* gn2_w    = (const float*)d_in[pi++];
  const float* gn2_b    = (const float*)d_in[pi++];
  const float* conv3_w  = (const float*)d_in[pi++];
  const float* conv3_b  = (const float*)d_in[pi++];

  const int Bn = 4, N = 4096, M = 4096, HID = 256, T = 1024, Hn = 8, dh = 32;

  // ---- workspace layout (floats), with region reuse ----
  float* ws = (float*)d_ws;
  size_t off = 0;
  float* feats = ws + off; off += (size_t)Bn * N * 36;          // 589824
  float* xhid  = ws + off; off += (size_t)16384 * 256;          // also qkv [4096,768]
  float* xgrid = ws + off; off += (size_t)Bn * M * 256;         // also x2 [B,256,64,64]
  float* xp    = ws + off; off += (size_t)4096 * 1024;          // also hmid / y
  float* xtok  = ws + off; off += (size_t)4096 * 256;           // DiT residual stream
  float* hbuf  = ws + off; off += (size_t)4096 * 256;           // h1 / attno / h2 / xf
  int*   idx   = (int*)(ws + off); off += (size_t)Bn * M * 3;
  float* wts   = ws + off; off += (size_t)Bn * M * 3;
  float* sc    = ws + off; off += (size_t)Bn * 256;             // silu(c)
  float* mod   = ws + off; off += (size_t)Bn * 1536;
  float* fmod  = ws + off; off += (size_t)Bn * 512;
  float* z1    = ws + off; off += (size_t)Bn * 256 * 32 * 32;
  float* z1g   = ws + off; off += (size_t)Bn * 256 * 32 * 32;
  float* z2    = ws + off; off += (size_t)Bn * 256 * 16 * 16;
  float* z2g   = ws + off; off += (size_t)Bn * 256 * 16 * 16;
  float* qkv   = xhid;   // reuse after gather
  float* hmid  = xp;     // reuse after patch GEMM
  float* yfin  = xp;
  float* x2    = xgrid;  // reuse after patchify

  // ---- 1. FFT features + input projection ----
  fft_feats_kernel<<<dim3(N / 256, Bn), 256, 0, stream>>>(x_noisy, feats, N);
  gemm_kernel<EPI_NONE><<<gemm_grid(Bn * N, HID), 128, 0, stream>>>(
      feats, in_w, in_b, xhid, Bn * N, HID, 36, nullptr, nullptr, 0, 1);

  // ---- 2. IDW interpolation onto latent grid ----
  knn_kernel<<<dim3(M / 256, Bn), 256, 0, stream>>>(coords, idx, wts, M, N);
  gather_kernel<<<Bn * M, 256, 0, stream>>>(xhid, idx, wts, xgrid, M, N, HID);

  // ---- 3. timestep conditioning ----
  temb_kernel<<<Bn, 256, 0, stream>>>(t_in, t_w1, t_b1, t_w2, t_b2, sc);

  // ---- 4. patchify + patch embed (+pos-embed epilogue) ----
  patchify_kernel<<<Bn * 1024, 256, 0, stream>>>(xgrid, xp);
  gemm_kernel<EPI_POSEMB><<<gemm_grid(Bn * T, HID), 128, 0, stream>>>(
      xp, patch_w, patch_b, xtok, Bn * T, HID, 1024, nullptr, nullptr, 0, T);

  // ---- 5. DiT blocks ----
  for (int i = 0; i < 4; ++i) {
    matvec_kernel<<<dim3(6, Bn), 256, 0, stream>>>(sc, blk[i].ada_w, blk[i].ada_b, mod, HID, 1536);
    // attn branch
    ln_mod_kernel<<<Bn * T, 256, 0, stream>>>(xtok, hbuf, mod, 1536, 256, 0, 10);
    gemm_kernel<EPI_NONE><<<gemm_grid(Bn * T, 768), 128, 0, stream>>>(
        hbuf, blk[i].qkv_w, blk[i].qkv_b, qkv, Bn * T, 768, HID, nullptr, nullptr, 0, 1);
    attn_kernel<<<dim3(T / 128, Bn * Hn), 256, 0, stream>>>(qkv, hbuf, Bn, T, Hn, dh);
    gemm_kernel<EPI_RESID_GATE><<<gemm_grid(Bn * T, HID), 128, 0, stream>>>(
        hbuf, blk[i].proj_w, blk[i].proj_b, xtok, Bn * T, HID, HID, xtok, mod + 512, 1536, T);
    // mlp branch
    ln_mod_kernel<<<Bn * T, 256, 0, stream>>>(xtok, hbuf, mod, 1536, 1024, 768, 10);
    gemm_kernel<EPI_GELU><<<gemm_grid(Bn * T, 1024), 128, 0, stream>>>(
        hbuf, blk[i].fc1_w, blk[i].fc1_b, hmid, Bn * T, 1024, HID, nullptr, nullptr, 0, 1);
    gemm_kernel<EPI_RESID_GATE><<<gemm_grid(Bn * T, HID), 128, 0, stream>>>(
        hmid, blk[i].fc2_w, blk[i].fc2_b, xtok, Bn * T, HID, 1024, xtok, mod + 1280, 1536, T);
  }

  // ---- 6. final layer + unpatchify ----
  matvec_kernel<<<dim3(2, Bn), 256, 0, stream>>>(sc, fl_ada_w, fl_ada_b, fmod, HID, 512);
  ln_mod_kernel<<<Bn * T, 256, 0, stream>>>(xtok, hbuf, fmod, 512, 256, 0, 10);
  gemm_kernel<EPI_NONE><<<gemm_grid(Bn * T, 1024), 128, 0, stream>>>(
      hbuf, fl_w, fl_b, yfin, Bn * T, 1024, HID, nullptr, nullptr, 0, 1);
  unpatchify_kernel<<<Bn * 1024, 256, 0, stream>>>(yfin, x2);

  // ---- 7. CNN downsample ----
  conv3x3_kernel<<<gemm_grid(Bn * 32 * 32, 256), 128, 0, stream>>>(
      x2, conv1_w, conv1_b, z1, Bn, 256, 64, 64, 256, 32, 32, 2, 0);
  gn_gelu_kernel<<<Bn * 8, 256, 0, stream>>>(z1, z1g, gn1_w, gn1_b, 256, 32 * 32, 8);
  conv3x3_kernel<<<gemm_grid(Bn * 16 * 16, 256), 128, 0, stream>>>(
      z1g, conv2_w, conv2_b, z2, Bn, 256, 32, 32, 256, 16, 16, 2, 0);
  gn_gelu_kernel<<<Bn * 8, 256, 0, stream>>>(z2, z2g, gn2_w, gn2_b, 256, 16 * 16, 8);
  conv3x3_kernel<<<gemm_grid(Bn * 16 * 16, 256), 128, 0, stream>>>(
      z2g, conv3_w, conv3_b, (float*)d_out, Bn, 256, 16, 16, 256, 16, 16, 1, 1);
}
